// MaskHeadSmallConv_16355235463839
// MI455X (gfx1250) — matrix operand order repaired and checked by reading, not compiled, not run on hardware
//
#include <hip/hip_runtime.h>
#include <hip/hip_bf16.h>
#include <math.h>

// ---------------------------------------------------------------------------
// Types for WMMA fragments (gfx1250, wave32)
// ---------------------------------------------------------------------------
typedef __attribute__((ext_vector_type(16))) _Float16 v16h;
typedef __attribute__((ext_vector_type(8)))  float    v8f;

#define GN_EPS 1e-05f
#define GN_GROUPS 2

// ---------------------------------------------------------------------------
// Kernel 1: concat(expand(x, Q), bbox_mask)  ->  h0 (32, 264, 20, 20)
// ---------------------------------------------------------------------------
__global__ void concat_kernel(const float* __restrict__ x,     // (2, 256, 20, 20)
                              const float* __restrict__ mask,  // (2, 16, 8, 20, 20)
                              float* __restrict__ out,         // (32, 264, 20, 20)
                              unsigned HID, unsigned NH, unsigned Q, unsigned HW,
                              unsigned total) {
    unsigned idx = blockIdx.x * blockDim.x + threadIdx.x;
    unsigned stride = gridDim.x * blockDim.x;
    unsigned C = HID + NH;
    for (; idx < total; idx += stride) {
        unsigned hw = idx % HW;
        unsigned c  = (idx / HW) % C;
        unsigned n  = idx / (HW * C);
        unsigned b  = n / Q;
        unsigned q  = n % Q;
        float v;
        if (c < HID) {
            v = x[(b * HID + c) * HW + hw];
        } else {
            v = mask[(((b * Q + q) * NH) + (c - HID)) * HW + hw];
        }
        out[idx] = v;
    }
}

// ---------------------------------------------------------------------------
// Kernel 2a: NCHW f32 -> NHWC f16 staging (channels zero-padded to Cpad)
// ---------------------------------------------------------------------------
__global__ void nhwc_f16_kernel(const float* __restrict__ in, _Float16* __restrict__ out,
                                unsigned C, unsigned Cpad, unsigned HW, unsigned total) {
    unsigned idx = blockIdx.x * blockDim.x + threadIdx.x;
    unsigned stride = gridDim.x * blockDim.x;
    for (; idx < total; idx += stride) {
        unsigned c  = idx % Cpad;
        unsigned s  = idx / Cpad;          // n*HW + hw
        unsigned n  = s / HW;
        unsigned hw = s - n * HW;
        _Float16 v = (_Float16)0.0f;
        if (c < C)
            v = (_Float16)in[(n * C + c) * HW + hw];
        out[idx] = v;
    }
}

// ---------------------------------------------------------------------------
// Kernel 2b: pack conv weights (Cout,Cin,KH,KW) f32 into A-fragment layout f16.
//   kk = j + 16*(lane/16)   (contiguous k order, matches B fragment)
//   k  = chunk*32 + kk ; tap = k / Cpad ; ci = k % Cpad ; m = tile*16 + lane%16
// ---------------------------------------------------------------------------
__global__ void wpack_kernel(const float* __restrict__ w, _Float16* __restrict__ out,
                             unsigned Cout, unsigned Cin, unsigned KH, unsigned KW,
                             unsigned Cpad, unsigned totalOut) {
    unsigned idx = blockIdx.x * blockDim.x + threadIdx.x;
    unsigned stride = gridDim.x * blockDim.x;
    const unsigned chunks = KH * KW * (Cpad >> 5);
    for (; idx < totalOut; idx += stride) {
        unsigned j    = idx & 15;
        unsigned lane = (idx >> 4) & 31;
        unsigned tc   = idx >> 9;                 // tile*chunks + chunk
        unsigned tile = tc / chunks;
        unsigned chunk = tc - tile * chunks;
        unsigned half = lane >> 4, lid = lane & 15;
        unsigned v = j >> 1, e = j & 1;
        unsigned kk = j + 16 * half; (void)v; (void)e;
        unsigned k  = chunk * 32 + kk;
        unsigned tap = k / Cpad;
        unsigned ci  = k - tap * Cpad;
        unsigned kh  = tap / KW;
        unsigned kw  = tap - kh * KW;
        unsigned m   = tile * 16 + lid;
        float val = 0.0f;
        if (m < Cout && ci < Cin)
            val = w[((m * Cin + ci) * KH + kh) * KW + kw];
        out[idx] = (_Float16)val;
    }
}

// ---------------------------------------------------------------------------
// Kernel 2c: implicit-GEMM conv via V_WMMA_F32_16X16X32_F16.
// Each wave computes a 16(Cout) x 32(spatial) tile: one shared A fragment,
// two B fragments, two accumulators per K chunk (2 WMMAs / A load).
// 4 waves/block -> block covers 128 spatial columns.
// All guards are value-level; EXEC is all-1s at every WMMA.
// ---------------------------------------------------------------------------
__global__ void conv_wmma_kernel(const _Float16* __restrict__ in_nhwc,
                                 const _Float16* __restrict__ wpack,
                                 const float* __restrict__ bias,
                                 float* __restrict__ out,
                                 int Cpad, int H, int W,
                                 int Cout, int KH, int KW, int pad) {
    const int lane = threadIdx.x & 31;
    const int wave = threadIdx.x >> 5;
    const int half = lane >> 4;
    const int lid  = lane & 15;

    const int mTile = blockIdx.x;
    const unsigned HW = (unsigned)(H * W);
    const unsigned sBase = (blockIdx.y * 4u + (unsigned)wave) * 32u;
    const unsigned s0 = sBase + (unsigned)lid;
    const unsigned s1 = s0 + 16u;

    // 32-bit spatial decode (all totals < 2^31)
    const int xw0 = (int)(s0 % (unsigned)W);
    const int yh0 = (int)((s0 / (unsigned)W) % (unsigned)H);
    const int nn0 = (int)(s0 / HW);
    const int xw1 = (int)(s1 % (unsigned)W);
    const int yh1 = (int)((s1 / (unsigned)W) % (unsigned)H);
    const int nn1 = (int)(s1 / HW);

    const int chunksPerTap = Cpad >> 5;
    const int totalChunks  = KH * KW * chunksPerTap;

    const v16h* __restrict__ A =
        ((const v16h*)wpack) + (long)mTile * totalChunks * 32 + lane;

    v8f acc0{};
    v8f acc1{};
    int chunk = 0;
    for (int kh = 0; kh < KH; ++kh) {
        const int iy0 = yh0 + kh - pad;
        const int iy1 = yh1 + kh - pad;
        for (int kw = 0; kw < KW; ++kw) {
            const int ix0 = xw0 + kw - pad;
            const int ix1 = xw1 + kw - pad;
            const bool inb0 = ((unsigned)iy0 < (unsigned)H) && ((unsigned)ix0 < (unsigned)W);
            const bool inb1 = ((unsigned)iy1 < (unsigned)H) && ((unsigned)ix1 < (unsigned)W);
            const long brow0 = (((long)nn0 * H + iy0) * W + ix0) * (long)Cpad + 16 * half;
            const long brow1 = (((long)nn1 * H + iy1) * W + ix1) * (long)Cpad + 16 * half;
            for (int c0 = 0; c0 < Cpad; c0 += 32) {
                v16h a = A[(long)chunk * 32];
                __builtin_prefetch((const void*)(A + (long)(chunk + 1) * 32), 0, 0);
                v16h b0{};
                v16h b1{};
                if (inb0) b0 = *(const v16h*)(in_nhwc + brow0 + c0);
                if (inb1) b1 = *(const v16h*)(in_nhwc + brow1 + c0);
                acc0 = __builtin_amdgcn_wmma_f32_16x16x32_f16(
                    false, a, false, b0, (short)0, acc0, false, false);
                acc1 = __builtin_amdgcn_wmma_f32_16x16x32_f16(
                    false, a, false, b1, (short)0, acc1, false, false);
                ++chunk;
            }
        }
    }

#pragma unroll
    for (int v = 0; v < 8; ++v) {
        int mo = mTile * 16 + v + 8 * half;
        if (mo < Cout) {
            float bv = bias[mo];
            out[(((long)nn0 * Cout + mo) * H + yh0) * W + xw0] = acc0[v] + bv;
            out[(((long)nn1 * Cout + mo) * H + yh1) * W + xw1] = acc1[v] + bv;
        }
    }
}

// ---------------------------------------------------------------------------
// Kernel 3a: GroupNorm statistics: mean & rsqrt(var+eps) per (n, group)
// ---------------------------------------------------------------------------
__global__ void gn_stats_kernel(const float* __restrict__ x, float* __restrict__ stats,
                                unsigned C, unsigned HW) {
    __shared__ float ssum[256];
    __shared__ float ssq[256];
    const unsigned ng = blockIdx.x;
    const unsigned g  = ng % GN_GROUPS;
    const unsigned n  = ng / GN_GROUPS;
    const unsigned cpg = C / GN_GROUPS;
    const unsigned base = (n * C + g * cpg) * HW;
    const unsigned cnt  = cpg * HW;

    float s = 0.0f, ss = 0.0f;
    for (unsigned i = threadIdx.x; i < cnt; i += blockDim.x) {
        float v = x[base + i];
        s += v;
        ss += v * v;
    }
    ssum[threadIdx.x] = s;
    ssq[threadIdx.x]  = ss;
    __syncthreads();
    for (int off = 128; off > 0; off >>= 1) {
        if ((int)threadIdx.x < off) {
            ssum[threadIdx.x] += ssum[threadIdx.x + off];
            ssq[threadIdx.x]  += ssq[threadIdx.x + off];
        }
        __syncthreads();
    }
    if (threadIdx.x == 0) {
        float mean = ssum[0] / (float)cnt;
        float var  = ssq[0] / (float)cnt - mean * mean;
        stats[2 * ng]     = mean;
        stats[2 * ng + 1] = rsqrtf(var + GN_EPS);
    }
}

// ---------------------------------------------------------------------------
// Kernel 3b: GroupNorm apply (+ optional ReLU), in place
// ---------------------------------------------------------------------------
__global__ void gn_apply_kernel(float* __restrict__ x, const float* __restrict__ stats,
                                const float* __restrict__ scale, const float* __restrict__ bias,
                                unsigned C, unsigned HW, unsigned total, int do_relu) {
    unsigned idx = blockIdx.x * blockDim.x + threadIdx.x;
    unsigned stride = gridDim.x * blockDim.x;
    const unsigned cpg = C / GN_GROUPS;
    for (; idx < total; idx += stride) {
        unsigned c = (idx / HW) % C;
        unsigned n = idx / (HW * C);
        unsigned ng = n * GN_GROUPS + c / cpg;
        float mean = stats[2 * ng];
        float istd = stats[2 * ng + 1];
        float v = (x[idx] - mean) * istd * scale[c] + bias[c];
        if (do_relu) v = fmaxf(v, 0.0f);
        x[idx] = v;
    }
}

// ---------------------------------------------------------------------------
// Kernel 4: out = expand(f) + upsample2x_nearest(h)
// ---------------------------------------------------------------------------
__global__ void upsample_add_kernel(const float* __restrict__ f, const float* __restrict__ h,
                                    float* __restrict__ out,
                                    unsigned C, unsigned Ho, unsigned Wo, unsigned rep,
                                    unsigned total) {
    unsigned idx = blockIdx.x * blockDim.x + threadIdx.x;
    unsigned stride = gridDim.x * blockDim.x;
    const unsigned Hi = Ho >> 1, Wi = Wo >> 1;
    for (; idx < total; idx += stride) {
        unsigned xw = idx % Wo;
        unsigned yh = (idx / Wo) % Ho;
        unsigned c  = (idx / (Wo * Ho)) % C;
        unsigned n  = idx / (Wo * Ho * C);
        float fv = f[(((n / rep) * C + c) * Ho + yh) * Wo + xw];
        float hv = h[((n * C + c) * Hi + (yh >> 1)) * Wi + (xw >> 1)];
        out[idx] = fv + hv;
    }
}

// ---------------------------------------------------------------------------
// Kernel 5: deformable conv 3x3 (Cin=32 -> Cout=16), bilinear w/ valid masks
// ---------------------------------------------------------------------------
#define DCN_COUT 16
__global__ void dcn_kernel(const float* __restrict__ x,     // (N, Cin, H, W)
                           const float* __restrict__ off,   // (N, 18, H, W)
                           const float* __restrict__ w,     // (16, Cin, 3, 3)
                           const float* __restrict__ b,     // (16)
                           float* __restrict__ out,         // (N, 16, H, W)
                           unsigned N, unsigned Cin, int H, int W) {
    unsigned idx = blockIdx.x * blockDim.x + threadIdx.x;
    unsigned total = N * (unsigned)(H * W);
    if (idx >= total) return;
    const unsigned HW = (unsigned)(H * W);
    int xw = (int)(idx % (unsigned)W);
    int yh = (int)((idx / (unsigned)W) % (unsigned)H);
    unsigned n = idx / HW;

    float acc[DCN_COUT];
#pragma unroll
    for (int co = 0; co < DCN_COUT; ++co) acc[co] = 0.0f;

    const unsigned offbase = n * 18u * HW + (unsigned)(yh * W + xw);

    for (int k = 0; k < 9; ++k) {
        const int ki = k / 3, kj = k % 3;
        float oy = off[offbase + (unsigned)(2 * k) * HW];
        float ox = off[offbase + (unsigned)(2 * k + 1) * HW];
        float py = (float)yh + (float)(ki - 1) + oy;
        float px = (float)xw + (float)(kj - 1) + ox;
        float fy0 = floorf(py), fx0 = floorf(px);
        float wy = py - fy0, wx = px - fx0;
        int y0 = (int)fy0, x0 = (int)fx0;
        int y1 = y0 + 1,   x1 = x0 + 1;

        float w00 = (1.0f - wy) * (1.0f - wx);
        float w01 = (1.0f - wy) * wx;
        float w10 = wy * (1.0f - wx);
        float w11 = wy * wx;

        bool vy0 = (y0 >= 0) && (y0 < H);
        bool vy1 = (y1 >= 0) && (y1 < H);
        bool vx0 = (x0 >= 0) && (x0 < W);
        bool vx1 = (x1 >= 0) && (x1 < W);
        int cy0 = min(max(y0, 0), H - 1);
        int cy1 = min(max(y1, 0), H - 1);
        int cx0 = min(max(x0, 0), W - 1);
        int cx1 = min(max(x1, 0), W - 1);

        float g00 = (vy0 && vx0) ? w00 : 0.0f;
        float g01 = (vy0 && vx1) ? w01 : 0.0f;
        float g10 = (vy1 && vx0) ? w10 : 0.0f;
        float g11 = (vy1 && vx1) ? w11 : 0.0f;

        unsigned i00 = (unsigned)(cy0 * W + cx0);
        unsigned i01 = (unsigned)(cy0 * W + cx1);
        unsigned i10 = (unsigned)(cy1 * W + cx0);
        unsigned i11 = (unsigned)(cy1 * W + cx1);

        for (unsigned ci = 0; ci < Cin; ++ci) {
            const unsigned base = (n * Cin + ci) * HW;
            float v = g00 * x[base + i00] + g01 * x[base + i01] +
                      g10 * x[base + i10] + g11 * x[base + i11];
            const float* wp = w + ci * 9 + (unsigned)k;
#pragma unroll
            for (int co = 0; co < DCN_COUT; ++co)
                acc[co] += v * wp[(unsigned)co * Cin * 9];
        }
    }
#pragma unroll
    for (int co = 0; co < DCN_COUT; ++co)
        out[(n * DCN_COUT + (unsigned)co) * HW + (unsigned)(yh * W + xw)] = acc[co] + b[co];
}

// ---------------------------------------------------------------------------
// Kernel 6: final 1x1 conv, 16 -> 1
// ---------------------------------------------------------------------------
__global__ void fin_kernel(const float* __restrict__ x, const float* __restrict__ w,
                           const float* __restrict__ b, float* __restrict__ out,
                           unsigned Cin, unsigned HW, unsigned total) {
    unsigned idx = blockIdx.x * blockDim.x + threadIdx.x;
    unsigned stride = gridDim.x * blockDim.x;
    for (; idx < total; idx += stride) {
        unsigned hw = idx % HW;
        unsigned n  = idx / HW;
        float s = b[0];
        for (unsigned c = 0; c < Cin; ++c)
            s += w[c] * x[(n * Cin + c) * HW + hw];
        out[idx] = s;
    }
}

// ---------------------------------------------------------------------------
// Host-side helpers
// ---------------------------------------------------------------------------
static inline int capgrid(long total, int tpb) {
    long b = (total + tpb - 1) / tpb;
    if (b > 65535) b = 65535;
    return (int)b;
}

// Stage input NCHW->NHWC(f16,pad) then run the WMMA conv.
static inline void run_conv(const float* in, const _Float16* wpacked, const float* bias,
                            float* out, _Float16* nhwc,
                            int N, int Cin, int H, int W, int Cout, int K, int pad,
                            hipStream_t stream) {
    int Cpad = ((Cin + 31) / 32) * 32;
    long NHW = (long)N * H * W;
    long stTotal = NHW * Cpad;
    nhwc_f16_kernel<<<capgrid(stTotal, 256), 256, 0, stream>>>(
        in, nhwc, (unsigned)Cin, (unsigned)Cpad, (unsigned)(H * W), (unsigned)stTotal);
    int mTiles = (Cout + 15) / 16;
    dim3 grid(mTiles, (int)(NHW / 128));   // NHW is a multiple of 128 everywhere here
    conv_wmma_kernel<<<grid, 128, 0, stream>>>(nhwc, wpacked, bias, out,
                                               Cpad, H, W, Cout, K, K, pad);
}

static inline long pack_weights(const float* w, _Float16* dst,
                                int Cout, int Cin, int K, hipStream_t stream) {
    int Cpad = ((Cin + 31) / 32) * 32;
    int mTiles = (Cout + 15) / 16;
    long chunks = (long)K * K * (Cpad >> 5);
    long totalOut = (long)mTiles * chunks * 32 * 16;   // f16 elements
    wpack_kernel<<<capgrid(totalOut, 256), 256, 0, stream>>>(
        w, dst, (unsigned)Cout, (unsigned)Cin, (unsigned)K, (unsigned)K,
        (unsigned)Cpad, (unsigned)totalOut);
    return totalOut;
}

static inline void launch_gn(float* x, const float* scale, const float* bias,
                             float* stats, int N, int C, int HW, int relu,
                             hipStream_t stream) {
    gn_stats_kernel<<<N * GN_GROUPS, 256, 0, stream>>>(x, stats, (unsigned)C, (unsigned)HW);
    long total = (long)N * C * HW;
    gn_apply_kernel<<<capgrid(total, 256), 256, 0, stream>>>(
        x, stats, scale, bias, (unsigned)C, (unsigned)HW, (unsigned)total, relu);
}

extern "C" void kernel_launch(void* const* d_in, const int* in_sizes, int n_in,
                              void* d_out, int out_size, void* d_ws, size_t ws_size,
                              hipStream_t stream) {
    // ---- problem constants (match reference setup_inputs) ----
    const int B = 2, Q = 16, NH = 8, HID = 256;
    const int H0 = 20, W0 = 20;
    const int NQ = B * Q;          // 32
    const int DIM = HID + NH;      // 264
    const int C2 = 128, C3 = 64, C4 = 32, C5 = 16;

    // ---- inputs ----
    const float* x        = (const float*)d_in[0];
    const float* bboxmask = (const float*)d_in[1];
    const float* fpn0     = (const float*)d_in[2];
    const float* fpn1     = (const float*)d_in[3];
    const float* fpn2     = (const float*)d_in[4];
    const float* lay1_w = (const float*)d_in[5];
    const float* lay1_b = (const float*)d_in[6];
    const float* gn1_s  = (const float*)d_in[7];
    const float* gn1_b  = (const float*)d_in[8];
    const float* lay2_w = (const float*)d_in[9];
    const float* lay2_b = (const float*)d_in[10];
    const float* gn2_s  = (const float*)d_in[11];
    const float* gn2_b  = (const float*)d_in[12];
    const float* lay3_w = (const float*)d_in[13];
    const float* lay3_b = (const float*)d_in[14];
    const float* gn3_s  = (const float*)d_in[15];
    const float* gn3_b  = (const float*)d_in[16];
    const float* lay4_w = (const float*)d_in[17];
    const float* lay4_b = (const float*)d_in[18];
    const float* gn4_s  = (const float*)d_in[19];
    const float* gn4_b  = (const float*)d_in[20];
    const float* off_w  = (const float*)d_in[21];
    const float* off_b  = (const float*)d_in[22];
    const float* dcn_w  = (const float*)d_in[23];
    const float* dcn_b  = (const float*)d_in[24];
    const float* gn5_s  = (const float*)d_in[25];
    const float* gn5_b  = (const float*)d_in[26];
    const float* fin_w  = (const float*)d_in[27];
    const float* fin_b  = (const float*)d_in[28];
    const float* ad1_w  = (const float*)d_in[29];
    const float* ad1_b  = (const float*)d_in[30];
    const float* ad2_w  = (const float*)d_in[31];
    const float* ad2_b  = (const float*)d_in[32];
    const float* ad3_w  = (const float*)d_in[33];
    const float* ad3_b  = (const float*)d_in[34];
    (void)in_sizes; (void)n_in; (void)ws_size; (void)out_size;

    // ---- workspace layout (float offsets), ~280 MB total ----
    float* ws = (float*)d_ws;
    float*     A0    = ws;                          // cap 14,745,600 floats
    float*     A1    = ws + 14745600L;              // cap 26,214,400 floats
    float*     FS    = ws + 40960000L;              // cap  1,638,400 floats
    float*     DS    = ws + 42598400L;              // cap 13,107,200 floats
    float*     STATS = ws + 55705600L;              // 256 floats
    _Float16*  NHWC  = (_Float16*)(ws + 55705856L); // cap 26,214,400 f16
    _Float16*  WP    = (_Float16*)(ws + 68813056L); // cap  2,097,152 f16

    // ---- pack all conv weights into A-fragment layout (f16) ----
    _Float16* wp_lay1 = WP;                long o = pack_weights(lay1_w, wp_lay1, DIM, DIM, 3, stream);
    _Float16* wp_lay2 = wp_lay1 + o;       o = pack_weights(lay2_w, wp_lay2, C2, DIM, 3, stream);
    _Float16* wp_lay3 = wp_lay2 + o;       o = pack_weights(lay3_w, wp_lay3, C3, C2, 3, stream);
    _Float16* wp_lay4 = wp_lay3 + o;       o = pack_weights(lay4_w, wp_lay4, C4, C3, 3, stream);
    _Float16* wp_ad1  = wp_lay4 + o;       o = pack_weights(ad1_w, wp_ad1, C2, 1024, 1, stream);
    _Float16* wp_ad2  = wp_ad1 + o;        o = pack_weights(ad2_w, wp_ad2, C3, 512, 1, stream);
    _Float16* wp_ad3  = wp_ad2 + o;        o = pack_weights(ad3_w, wp_ad3, C4, 256, 1, stream);
    _Float16* wp_off  = wp_ad3 + o;        (void)pack_weights(off_w, wp_off, 18, C4, 1, stream);

    // ---- 1) h0 = concat(expand(x,Q), bbox_mask): (32,264,20,20) -> A0 ----
    {
        long total = (long)NQ * DIM * H0 * W0;
        concat_kernel<<<capgrid(total, 256), 256, 0, stream>>>(
            x, bboxmask, A0, (unsigned)HID, (unsigned)NH, (unsigned)Q,
            (unsigned)(H0 * W0), (unsigned)total);
    }

    // ---- 2) lay1: 3x3 264->264, GN, ReLU -> A1 ----
    run_conv(A0, wp_lay1, lay1_b, A1, NHWC, NQ, DIM, H0, W0, DIM, 3, 1, stream);
    launch_gn(A1, gn1_s, gn1_b, STATS, NQ, DIM, H0 * W0, 1, stream);

    // ---- 3) lay2: 3x3 264->128, GN, ReLU -> A0 ----
    run_conv(A1, wp_lay2, lay2_b, A0, NHWC, NQ, DIM, H0, W0, C2, 3, 1, stream);
    launch_gn(A0, gn2_s, gn2_b, STATS, NQ, C2, H0 * W0, 1, stream);

    // ---- 4) ad1: 1x1 1024->128 on fpn0 (2,1024,40,40) -> FS ----
    run_conv(fpn0, wp_ad1, ad1_b, FS, NHWC, B, 1024, 2 * H0, 2 * W0, C2, 1, 0, stream);

    // ---- 5) h3 = expand(f1) + up2(h2): (32,128,40,40) -> A1 ----
    {
        long total = (long)NQ * C2 * (2 * H0) * (2 * W0);
        upsample_add_kernel<<<capgrid(total, 256), 256, 0, stream>>>(
            FS, A0, A1, (unsigned)C2, (unsigned)(2 * H0), (unsigned)(2 * W0),
            (unsigned)Q, (unsigned)total);
    }

    // ---- 6) lay3: 3x3 128->64, GN, ReLU -> A0 ----
    run_conv(A1, wp_lay3, lay3_b, A0, NHWC, NQ, C2, 2 * H0, 2 * W0, C3, 3, 1, stream);
    launch_gn(A0, gn3_s, gn3_b, STATS, NQ, C3, 4 * H0 * W0, 1, stream);

    // ---- 7) ad2: 1x1 512->64 on fpn1 (2,512,80,80) -> FS ----
    run_conv(fpn1, wp_ad2, ad2_b, FS, NHWC, B, 512, 4 * H0, 4 * W0, C3, 1, 0, stream);

    // ---- 8) h5 = expand(f2) + up2(h4): (32,64,80,80) -> A1 ----
    {
        long total = (long)NQ * C3 * (4 * H0) * (4 * W0);
        upsample_add_kernel<<<capgrid(total, 256), 256, 0, stream>>>(
            FS, A0, A1, (unsigned)C3, (unsigned)(4 * H0), (unsigned)(4 * W0),
            (unsigned)Q, (unsigned)total);
    }

    // ---- 9) lay4: 3x3 64->32, GN, ReLU -> A0 ----
    run_conv(A1, wp_lay4, lay4_b, A0, NHWC, NQ, C3, 4 * H0, 4 * W0, C4, 3, 1, stream);
    launch_gn(A0, gn4_s, gn4_b, STATS, NQ, C4, 16 * H0 * W0, 1, stream);

    // ---- 10) ad3: 1x1 256->32 on fpn2 (2,256,160,160) -> FS ----
    run_conv(fpn2, wp_ad3, ad3_b, FS, NHWC, B, 256, 8 * H0, 8 * W0, C4, 1, 0, stream);

    // ---- 11) h7 = expand(f3) + up2(h6): (32,32,160,160) -> A1 ----
    {
        long total = (long)NQ * C4 * (8 * H0) * (8 * W0);
        upsample_add_kernel<<<capgrid(total, 256), 256, 0, stream>>>(
            FS, A0, A1, (unsigned)C4, (unsigned)(8 * H0), (unsigned)(8 * W0),
            (unsigned)Q, (unsigned)total);
    }

    // ---- 12) offset = 1x1 conv 32->18 on h7 -> A0 ----
    run_conv(A1, wp_off, off_b, A0, NHWC, NQ, C4, 8 * H0, 8 * W0, 18, 1, 0, stream);

    // ---- 13) deformable conv 3x3 32->16: (h7, offset) -> DS ----
    {
        long total = (long)NQ * (8 * H0) * (8 * W0);
        dcn_kernel<<<(int)((total + 255) / 256), 256, 0, stream>>>(
            A1, A0, dcn_w, dcn_b, DS, (unsigned)NQ, (unsigned)C4, 8 * H0, 8 * W0);
    }

    // ---- 14) gn5 + ReLU in place on DS ----
    launch_gn(DS, gn5_s, gn5_b, STATS, NQ, C5, 64 * H0 * W0, 1, stream);

    // ---- 15) fin: 1x1 conv 16->1 -> d_out ----
    {
        long total = (long)NQ * (8 * H0) * (8 * W0);
        fin_kernel<<<capgrid(total, 256), 256, 0, stream>>>(
            DS, fin_w, fin_b, (float*)d_out, (unsigned)C5, (unsigned)(64 * H0 * W0),
            (unsigned)total);
    }
}
